// PropagationBlock_57329223467239
// MI455X (gfx1250) — matrix operand order repaired and checked by reading, not compiled, not run on hardware
//
#include <hip/hip_runtime.h>
#include <hip/hip_bf16.h>

// ---------------------------------------------------------------------------
// PropagationBlock for MI455X (gfx1250, wave32, WMMA).
// Heavy per-edge matmuls use V_WMMA_F32_16X16X32_BF16 (bf16 in, fp32 acc).
// Node array (19.2MB) is L2-resident (192MB L2); kernel is L2/atomic bound,
// HBM traffic ~150MB << 23.3 TB/s capability.  Scatter address math is
// precomputed into a packed per-row LDS table so the hot epilogue is
// ds_load_b64 + unpack + global_atomic_add_f32 (SADDR + 32-bit voffset).
// ---------------------------------------------------------------------------

#define N_NODES 50000
#define N_EDGES 1000000
#define NV 32

typedef __attribute__((ext_vector_type(4)))  float  f32x4;
typedef __attribute__((ext_vector_type(8)))  float  v8f;
typedef __attribute__((ext_vector_type(16))) __bf16 v16bf;

union Frag16 {            // 32 bytes = 8 VGPRs: one 16-bit WMMA A/B fragment
    v16bf bf;
    f32x4 f[2];
};

__device__ __forceinline__ unsigned short tobf(float x) {
    union { float f; unsigned u; } a; a.f = x;
    unsigned r = a.u + 0x7FFFu + ((a.u >> 16) & 1u);   // round-to-nearest-even
    return (unsigned short)(r >> 16);
}
__device__ __forceinline__ float frombf(unsigned short h) {
    union { unsigned u; float f; } t; t.u = ((unsigned)h) << 16; return t.f;
}
__device__ __forceinline__ float silu_f(float z) {
    return z / (1.0f + __expf(-z));
}

// ---------------------------------------------------------------------------
// pb_prep: fold the edge-side mixing matrices.
//   Mv = (vv1+vv2)/2 ; CA = 0.5*n2e1@Mv ; CB = 0.25*n2e2@Mv
//   P  = CA+CB (applied to W1*x_src) ; Q = CB-CA (applied to W1*x_dst)
//   R  = (e2n1+e2n2)/2 (applied to acc1) ; S = (e2n2-e2n1)/2 (applied to acc2)
// ---------------------------------------------------------------------------
__global__ __launch_bounds__(1024) void pb_prep(
    const float* __restrict__ n2e1, const float* __restrict__ n2e2,
    const float* __restrict__ vv1,  const float* __restrict__ vv2,
    const float* __restrict__ e2n1, const float* __restrict__ e2n2,
    float* __restrict__ P, float* __restrict__ Q,
    float* __restrict__ R, float* __restrict__ S)
{
    __shared__ float sMv[32 * 32];
    const int t = threadIdx.x;
    const int r = t >> 5, c = t & 31;
    sMv[t] = 0.5f * (vv1[t] + vv2[t]);
    __syncthreads();
    float ca = 0.f, cb = 0.f;
#pragma unroll
    for (int k = 0; k < 32; ++k) {
        float m = sMv[k * 32 + c];
        ca += n2e1[r * 32 + k] * m;
        cb += n2e2[r * 32 + k] * m;
    }
    ca *= 0.5f; cb *= 0.25f;
    P[t] = ca + cb;
    Q[t] = cb - ca;
    R[t] = 0.5f * (e2n1[t] + e2n2[t]);
    S[t] = 0.5f * (e2n2[t] - e2n1[t]);
}

// ---------------------------------------------------------------------------
__global__ void pb_zero(float* __restrict__ p, int n4)
{
    int i = blockIdx.x * blockDim.x + threadIdx.x;
    if (i < n4) {
        f32x4 z = {};
        reinterpret_cast<f32x4*>(p)[i] = z;
    }
}

// ---------------------------------------------------------------------------
// pb_node_pre: MixVecScalar.  One wave per node; lane = channel.
// ---------------------------------------------------------------------------
__global__ __launch_bounds__(256) void pb_node_pre(
    const float* __restrict__ xn, const float* __restrict__ xn_attr,
    const float* __restrict__ msM, const float* __restrict__ msW,
    float* __restrict__ xn2)
{
    __shared__ float shM[32 * 32];
    __shared__ float shA[8][96];
    __shared__ float shT[8][96];

    const int tid  = threadIdx.x;
    const int wave = tid >> 5;
    const int lane = tid & 31;
    for (int i = tid; i < 1024; i += 256) shM[i] = msM[i];

    const int node = blockIdx.x * 8 + wave;
    const bool ok  = node < N_NODES;

    float a0 = 0.f, a1 = 0.f, a2 = 0.f, at = 0.f;
    if (ok) {
        const float* p = xn + (size_t)node * 96;
        a0 = p[lane]; a1 = p[32 + lane]; a2 = p[64 + lane];
        at = xn_attr[(size_t)node * 32 + lane];
        shA[wave][lane] = a0; shA[wave][32 + lane] = a1; shA[wave][64 + lane] = a2;
    }
    __syncthreads();
    if (ok) {
        float t0 = 0.f, t1 = 0.f, t2 = 0.f;
#pragma unroll
        for (int v = 0; v < 32; ++v) {
            float m = shM[v * 32 + lane];
            t0 += shA[wave][v]      * m;
            t1 += shA[wave][32 + v] * m;
            t2 += shA[wave][64 + v] * m;
        }
        shT[wave][lane] = t0 * at; shT[wave][32 + lane] = t1 * at; shT[wave][64 + lane] = t2 * at;
    }
    __syncthreads();
    if (ok) {
        float x0 = 0.f, x1 = 0.f, x2 = 0.f;
#pragma unroll
        for (int s = 0; s < 32; ++s) {
            float m = shM[lane * 32 + s];
            x0 += shT[wave][s]      * m;
            x1 += shT[wave][32 + s] * m;
            x2 += shT[wave][64 + s] * m;
        }
        float ang = 0.1f * msW[0];
        float ca = __cosf(ang), sa = __sinf(ang);
        float* o = xn2 + (size_t)node * 96;
        o[lane]      = ca * a0 + sa * x0;
        o[32 + lane] = ca * a1 + sa * x1;
        o[64 + lane] = ca * a2 + sa * x2;
    }
}

// ---------------------------------------------------------------------------
// pb_edge: the hot kernel.  2 waves/block, 32 edges/wave (96 rows = 6 M-tiles).
//  phase 1: stage u=W1*x_src, v=W1*x_dst (bf16) + W2 (bf16) in LDS; build a
//           packed per-row scatter table: u64 = {src_off, dst_off|edge_id}
//           (offsets are multiples of 32 floats -> low 5 bits carry edge id)
//  phase 2: per M-tile, per N-half: C = u@P + v@Q via two chained
//           v_wmma_f32_16x16x32_bf16, then scatter W2*C with f32 atomics
//           addressed as uniform-base + 32-bit offset.
// A-fragment layout (dense 16-bit A, ISA 7.12.2): lane<16 -> K runs {0-7,16-23},
// lane>=16 -> {8-15,24-31}; each run = 16B -> two ds_load_b128 per fragment.
// B-fragment layout (dense 32x16 16-bit B): lane<16 col=lane K={0-7,16-23},
// lane>=16 col=lane-16 K={8-15,24-31} -> column-major bf16 LDS, two b128 per frag.
// ---------------------------------------------------------------------------
#define NTILES (N_EDGES / 32)   // 31250
#define WPB 2                   // waves (32-edge tiles) per block

__global__ __launch_bounds__(32 * WPB) void pb_edge(
    const float* __restrict__ xn2,
    const float* __restrict__ xe_attr,
    const int*   __restrict__ xe_src,
    const int*   __restrict__ xe_dst,
    const float* __restrict__ fc1w, const float* __restrict__ fc1b,
    const float* __restrict__ fc2w, const float* __restrict__ fc2b,
    const float* __restrict__ Pm,   const float* __restrict__ Qm,
    float* __restrict__ acc1, float* __restrict__ acc2)
{
    __shared__ __align__(16) unsigned short shPT[32 * 32];       // P^T bf16 [n][k]
    __shared__ __align__(16) unsigned short shQT[32 * 32];       // Q^T bf16 [n][k]
    __shared__ __align__(16) unsigned short shU[WPB][96 * 32];   // u rows, bf16
    __shared__ __align__(16) unsigned short shV[WPB][96 * 32];   // v rows, bf16
    __shared__ unsigned short      shW2[WPB][32 * 32];           // W2 bf16
    __shared__ unsigned long long  shRowP[WPB][96];              // packed scatter table
    __shared__ int   shS[WPB][32];
    __shared__ int   shD[WPB][32];
    __shared__ float shAttr[WPB][32];

    const int tid  = threadIdx.x;
    const int wave = tid >> 5;
    const int lane = tid & 31;

    // transpose+quantize P,Q once per block
    for (int i = tid; i < 1024; i += 32 * WPB) {
        int n = i >> 5, k = i & 31;
        shPT[n * 32 + k] = tobf(Pm[k * 32 + n]);
        shQT[n * 32 + k] = tobf(Qm[k * 32 + n]);
    }

    // NTILES is an exact multiple of WPB -> every wave always has a full tile
    const int tile  = blockIdx.x * WPB + wave;
    const int ebase = tile * 32;

    shS[wave][lane]    = xe_src[ebase + lane];
    shD[wave][lane]    = xe_dst[ebase + lane];
    shAttr[wave][lane] = xe_attr[ebase + lane];
    __syncthreads();

    // packed scatter table: 3 rows per lane (the only div-by-3 in the kernel)
    for (int rr = lane; rr < 96; rr += 32) {
        const int el = rr / 3;
        const int d  = rr - el * 3;
        const unsigned o1 = (unsigned)(shD[wave][el] * 96 + d * 32) | (unsigned)el;
        const unsigned o2 = (unsigned)(shS[wave][el] * 96 + d * 32);
        shRowP[wave][rr] = ((unsigned long long)o2 << 32) | (unsigned long long)o1;
    }

    {   // stage u, v, W2
        const float f1w = fc1w[lane], f1b = fc1b[lane];
        const float f2w = fc2w[lane], f2b = fc2b[lane];
        for (int e = 0; e < 32; ++e) {
            const float a  = shAttr[wave][e];
            const float w1 = silu_f(a * f1w + f1b);
            const float w2 = silu_f(a * f2w + f2b);
            shW2[wave][e * 32 + lane] = tobf(w2);
            const float* ps = xn2 + (size_t)shS[wave][e] * 96;
            const float* pd = xn2 + (size_t)shD[wave][e] * 96;
#pragma unroll
            for (int d = 0; d < 3; ++d) {
                shU[wave][(e * 3 + d) * 32 + lane] = tobf(w1 * ps[d * 32 + lane]);
                shV[wave][(e * 3 + d) * 32 + lane] = tobf(w1 * pd[d * 32 + lane]);
            }
        }
    }
    __syncthreads();

    const int half = lane >> 4;
    const int m15  = lane & 15;

    // loop-invariant B fragments for P and Q (both N-halves)
    Frag16 bP[2], bQ[2];
#pragma unroll
    for (int j = 0; j < 2; ++j) {
        const int colg = m15 + 16 * j;
        const unsigned short* pp = shPT + colg * 32 + half * 8;
        bP[j].f[0] = *reinterpret_cast<const f32x4*>(pp);
        bP[j].f[1] = *reinterpret_cast<const f32x4*>(pp + 16);
        const unsigned short* qq = shQT + colg * 32 + half * 8;
        bQ[j].f[0] = *reinterpret_cast<const f32x4*>(qq);
        bQ[j].f[1] = *reinterpret_cast<const f32x4*>(qq + 16);
    }

    const int kb = half * 8;
    for (int mt = 0; mt < 6; ++mt) {
        Frag16 aU, aV;
        const unsigned short* ru = shU[wave] + (mt * 16 + m15) * 32;
        aU.f[0] = *reinterpret_cast<const f32x4*>(ru + kb);
        aU.f[1] = *reinterpret_cast<const f32x4*>(ru + 16 + kb);
        const unsigned short* rv = shV[wave] + (mt * 16 + m15) * 32;
        aV.f[0] = *reinterpret_cast<const f32x4*>(rv + kb);
        aV.f[1] = *reinterpret_cast<const f32x4*>(rv + 16 + kb);

        const int rbase = mt * 16 + 8 * half;       // first row this lane scatters

#pragma unroll
        for (int j = 0; j < 2; ++j) {
            v8f c = {};
            c = __builtin_amdgcn_wmma_f32_16x16x32_bf16(
                    false, aU.bf, false, bP[j].bf, (short)0, c, false, false);
            c = __builtin_amdgcn_wmma_f32_16x16x32_bf16(
                    false, aV.bf, false, bQ[j].bf, (short)0, c, false, false);

            const unsigned ncol = (unsigned)(m15 + 16 * j);
#pragma unroll
            for (int i = 0; i < 8; ++i) {
                const unsigned long long pk = shRowP[wave][rbase + i];  // b64, imm offset
                const unsigned lo = (unsigned)pk;            // dst_off | edge_id
                const unsigned o2 = (unsigned)(pk >> 32);    // src_off
                const float w2  = frombf(shW2[wave][((lo & 31u) << 5) + ncol]);
                const float val = w2 * c[i];
                atomicAdd(acc1 + ((lo & ~31u) + ncol), val); // dst segment sum
                atomicAdd(acc2 + (o2 + ncol), val);          // src segment sum
            }
        }
    }
}

// ---------------------------------------------------------------------------
// pb_node_post:  y = acc1@R + acc2@S ; out = y * tanh(||y||_dim)
// ---------------------------------------------------------------------------
__global__ __launch_bounds__(256) void pb_node_post(
    const float* __restrict__ acc1, const float* __restrict__ acc2,
    const float* __restrict__ Rm,   const float* __restrict__ Sm,
    float* __restrict__ out)
{
    __shared__ float shR[32 * 32];
    __shared__ float shSS[32 * 32];
    __shared__ float shX[8][96];
    __shared__ float shY[8][96];

    const int tid  = threadIdx.x;
    const int wave = tid >> 5;
    const int lane = tid & 31;
    for (int i = tid; i < 1024; i += 256) { shR[i] = Rm[i]; shSS[i] = Sm[i]; }

    const int node = blockIdx.x * 8 + wave;
    const bool ok  = node < N_NODES;
    if (ok) {
        const float* p1 = acc1 + (size_t)node * 96;
        const float* p2 = acc2 + (size_t)node * 96;
#pragma unroll
        for (int d = 0; d < 3; ++d) {
            shX[wave][d * 32 + lane] = p1[d * 32 + lane];
            shY[wave][d * 32 + lane] = p2[d * 32 + lane];
        }
    }
    __syncthreads();
    if (ok) {
        float y0 = 0.f, y1 = 0.f, y2 = 0.f;
#pragma unroll
        for (int v = 0; v < 32; ++v) {
            const float r = shR[v * 32 + lane];
            const float s = shSS[v * 32 + lane];
            y0 += shX[wave][v]      * r + shY[wave][v]      * s;
            y1 += shX[wave][32 + v] * r + shY[wave][32 + v] * s;
            y2 += shX[wave][64 + v] * r + shY[wave][64 + v] * s;
        }
        const float nrm = sqrtf(y0 * y0 + y1 * y1 + y2 * y2);
        const float th  = tanhf(nrm);
        float* o = out + (size_t)node * 96;
        o[lane]      = y0 * th;
        o[32 + lane] = y1 * th;
        o[64 + lane] = y2 * th;
    }
}

// ---------------------------------------------------------------------------
extern "C" void kernel_launch(void* const* d_in, const int* in_sizes, int n_in,
                              void* d_out, int out_size, void* d_ws, size_t ws_size,
                              hipStream_t stream)
{
    const float* xn      = (const float*)d_in[0];
    const float* xn_attr = (const float*)d_in[1];
    const float* xe_attr = (const float*)d_in[2];
    const int*   xe_src  = (const int*)  d_in[3];
    const int*   xe_dst  = (const int*)  d_in[4];
    const float* ms_M    = (const float*)d_in[5];
    const float* ms_w    = (const float*)d_in[6];
    const float* fc1_w   = (const float*)d_in[7];
    const float* fc1_b   = (const float*)d_in[8];
    const float* fc2_w   = (const float*)d_in[9];
    const float* fc2_b   = (const float*)d_in[10];
    const float* n2e_M1  = (const float*)d_in[11];
    const float* n2e_M2  = (const float*)d_in[12];
    const float* vv_M1   = (const float*)d_in[13];
    const float* vv_M2   = (const float*)d_in[14];
    const float* e2n_M1  = (const float*)d_in[15];
    const float* e2n_M2  = (const float*)d_in[16];

    // workspace layout (floats): xn2 | acc1 | acc2 | P | Q | R | S  (~57.6 MB)
    float* ws   = (float*)d_ws;
    const size_t NODE_F = (size_t)N_NODES * 96;   // 4,800,000
    float* xn2  = ws;
    float* acc1 = ws + NODE_F;
    float* acc2 = ws + 2 * NODE_F;
    float* Pm   = ws + 3 * NODE_F;
    float* Qm   = Pm + 1024;
    float* Rm   = Qm + 1024;
    float* Sm   = Rm + 1024;

    // 1) fold matrices
    pb_prep<<<dim3(1), dim3(1024), 0, stream>>>(n2e_M1, n2e_M2, vv_M1, vv_M2,
                                                e2n_M1, e2n_M2, Pm, Qm, Rm, Sm);
    // 2) zero both accumulators (2 * 4.8M floats = 2.4M float4)
    {
        const int n4 = (int)(2 * NODE_F / 4);
        pb_zero<<<dim3((n4 + 255) / 256), dim3(256), 0, stream>>>(acc1, n4);
    }
    // 3) node pre-mix
    pb_node_pre<<<dim3((N_NODES + 7) / 8), dim3(256), 0, stream>>>(
        xn, xn_attr, ms_M, ms_w, xn2);
    // 4) edge kernel (WMMA) — 31250 wave-tiles, 2 waves/block, exact cover
    pb_edge<<<dim3(NTILES / WPB), dim3(32 * WPB), 0, stream>>>(
        xn2, xe_attr, xe_src, xe_dst, fc1_w, fc1_b, fc2_w, fc2_b,
        Pm, Qm, acc1, acc2);
    // 5) node post-mix + activation -> d_out
    pb_node_post<<<dim3((N_NODES + 7) / 8), dim3(256), 0, stream>>>(
        acc1, acc2, Rm, Sm, (float*)d_out);
}
